// PixelCNN_47897475285189
// MI455X (gfx1250) — matrix operand order repaired and checked by reading, not compiled, not run on hardware
//
#include <hip/hip_runtime.h>
#include <hip/hip_bf16.h>
#include <math.h>
#include <stdint.h>

typedef __attribute__((ext_vector_type(16))) _Float16 v16h;
typedef __attribute__((ext_vector_type(8)))  _Float16 v8h;
typedef __attribute__((ext_vector_type(8)))  float    v8f;
typedef __attribute__((ext_vector_type(4)))  unsigned int u32x4;
typedef __attribute__((ext_vector_type(8)))  int i32x8;
typedef __attribute__((ext_vector_type(4)))  int i32x4;

#define NPIX 65536   // 16 * 64 * 64

// ---------------------------------------------------------------------------
// Tensor Data Mover: 2D tile load global->LDS.
//   rows x rowUnits (8-byte units per row), global row stride strideUnits.
// D# per CDNA5 ISA 8.3/8.4: group0 = {count|flags, lds_addr, global_addr,
// type=2}; group1 = {data_size, tensor dims, tile dims, dim0 stride}.
// This toolchain exposes the 6-arg builtin (extra descriptor group + cpol).
// ---------------------------------------------------------------------------
__device__ __forceinline__ void tdm_load_w(const _Float16* gsrc, unsigned ldsOff,
                                           unsigned rows, unsigned rowUnits,
                                           unsigned long long strideUnits)
{
  unsigned long long ga = (unsigned long long)(uintptr_t)gsrc;
  u32x4 g0;
  g0[0] = 1u;                                            // count=1, user mode
  g0[1] = ldsOff;                                        // lds_addr (bytes)
  g0[2] = (unsigned)(ga & 0xFFFFFFFFu);                  // global_addr[31:0]
  g0[3] = (unsigned)((ga >> 32) & 0x01FFFFFFu) | (2u << 30);  // [56:32] | type=2
  i32x8 g1;
  g1[0] = (int)(3u << 16);                               // data_size = 8 bytes
  g1[1] = (int)((rowUnits & 0xFFFFu) << 16);             // tensor_dim0[15:0]
  g1[2] = (int)((rowUnits >> 16) | ((rows & 0xFFFFu) << 16));   // dim0 hi | dim1 lo
  g1[3] = (int)((rows >> 16) | (rowUnits << 16));        // dim1 hi | tile_dim0
  g1[4] = (int)rows;                                     // tile_dim1 (tile_dim2=0)
  g1[5] = (int)(strideUnits & 0xFFFFFFFFull);            // dim0 stride lo
  g1[6] = (int)((strideUnits >> 32) & 0xFFFFull);        // dim0 stride hi
  g1[7] = 0;
  i32x4 z4 = {0, 0, 0, 0};
  i32x8 z8 = {0, 0, 0, 0, 0, 0, 0, 0};
  __builtin_amdgcn_tensor_load_to_lds(g0, g1, z4, z4, z8, 0);
}

// ---------------------------------------------------------------------------
// Input prep: normalize x, append ones channel, pad to 32 ch (f16),
// and build the row-shifted vertical-stack input vx.
// ---------------------------------------------------------------------------
__global__ void prep_input(const float* __restrict__ x,
                           _Float16* __restrict__ hx0,   // [NPIX][32]
                           _Float16* __restrict__ vx0)   // [NPIX][32]
{
  int p = blockIdx.x * blockDim.x + threadIdx.x;
  if (p >= NPIX) return;
  int nb = p >> 12;
  int hw = p & 4095;
  int h  = hw >> 6;
  int w  = hw & 63;

  _Float16 hv[32];
#pragma unroll
  for (int c = 0; c < 32; ++c) hv[c] = (_Float16)0.f;
  for (int c = 0; c < 3; ++c) {
    float v = x[((nb * 3 + c) << 12) + hw];
    hv[c] = (_Float16)((v / 127.5f - 1.f) * 2.f);
  }
  hv[3] = (_Float16)1.f;
  for (int c = 0; c < 32; ++c) hx0[(size_t)p * 32 + c] = hv[c];

  _Float16 vv[32];
#pragma unroll
  for (int c = 0; c < 32; ++c) vv[c] = (_Float16)0.f;
  if (h != 0) {
    int hwU = ((h - 1) << 6) + w;
    for (int c = 0; c < 3; ++c) {
      float v = x[((nb * 3 + c) << 12) + hwU];
      vv[c] = (_Float16)((v / 127.5f - 1.f) * 2.f);
    }
    vv[3] = (_Float16)1.f;
  }
  for (int c = 0; c < 32; ++c) vx0[(size_t)p * 32 + c] = vv[c];
}

// ---------------------------------------------------------------------------
// Pack fp32 weights [O][I][taps] into WMMA B-fragment order:
// [tap][kchunk][ntile][lane(32)][16 halves], applying the subpixel causal
// mask (0=none, 1='A', 2='B') computed on the REAL dims with C=3 groups.
// ---------------------------------------------------------------------------
__global__ void pack_w(const float* __restrict__ w, _Float16* __restrict__ dst,
                       int O, int I, int taps, int Kch, int NT, int maskMode)
{
  int gid = blockIdx.x * blockDim.x + threadIdx.x;
  int total = taps * Kch * NT * 32;
  if (gid >= total) return;
  int lane = gid & 31;
  int q = gid >> 5;
  int nt = q % NT; q /= NT;
  int kc = q % Kch; q /= Kch;
  int t  = q;
  int n = lane & 15;
  int o = nt * 16 + n;
  int kbase = kc * 32 + ((lane >> 4) << 4);
  v16h vals;
#pragma unroll
  for (int e = 0; e < 16; ++e) {
    int k = kbase + e;
    float v = 0.f;
    if (o < O && k < I) {
      v = w[((size_t)o * I + k) * taps + t];
      if (maskMode) {
        int isub = I / 3, osub = O / 3;
        int g = o / osub;
        int lim = isub * (g + (maskMode == 2 ? 1 : 0));
        if (k >= lim) v = 0.f;
      }
    }
    vals[e] = (_Float16)v;
  }
  *(v16h*)(dst + ((size_t)gid << 4)) = vals;
}

// ---------------------------------------------------------------------------
// Initialize the f32 accumulator: C[p][n] = b1[n] (+ b2[n]) (+ residual f16)
// ---------------------------------------------------------------------------
__global__ void init_c(float* __restrict__ C, const float* __restrict__ b1,
                       const float* __restrict__ b2,
                       const _Float16* __restrict__ add,
                       int Nreal, int Npitch)
{
  size_t idx = (size_t)blockIdx.x * blockDim.x + threadIdx.x;
  size_t total = (size_t)NPIX * Npitch;
  if (idx >= total) return;
  int n = (int)(idx % Npitch);
  size_t p = idx / Npitch;
  float v = 0.f;
  if (n < Nreal) {
    v = b1[n];
    if (b2) v += b2[n];
    if (add) v += (float)add[p * Npitch + n];
  }
  C[idx] = v;
}

// ---------------------------------------------------------------------------
// Implicit-GEMM tap kernel. One wave owns 2 M-tiles (32 pixels) x 16 outch.
// Weight B-fragments are streamed global->LDS with the Tensor Data Mover,
// double-buffered across taps (s_wait_tensorcnt pipelining). A tiles are
// staged in LDS (bounds-zeroed causal shifts) and read back in contiguous k order
// (k = 16*half + s), as B is packed. K-chunk loop fully unrolled (KCH template).
// ---------------------------------------------------------------------------
template<int KCH>
__global__ __launch_bounds__(128) void gemm_tap(
    const _Float16* __restrict__ A, int Apitch,
    const _Float16* __restrict__ W, int NT,
    int ntaps, int kw, int dyB, int dxB,
    float* __restrict__ C, int Npitch)
{
  __shared__ __align__(16) _Float16 sA[32 * 32];
  __shared__ __align__(16) _Float16 sW[2 * 4 * KCH * 512];  // 2 bufs x 4 waves
  const int tid  = threadIdx.x;
  const int lane = tid & 31;
  const int wave = tid >> 5;
  const int ntile = blockIdx.y * (blockDim.x >> 5) + wave;
  const bool active = ntile < NT;
  const int p0 = blockIdx.x << 5;          // 32 pixels, same batch & row
  const int h  = (p0 >> 6) & 63;
  const int w0 = p0 & 63;
  const int nbase = (p0 >> 12) << 12;
  const int l15 = lane & 15;
  const int mo  = (lane >> 4) << 3;        // +8 row offset for upper half-wave
  const unsigned wStride = (unsigned)NT * 128u;   // 8-byte units between kchunks

  v8f c0 = {}, c1 = {};
  if (active) {
#pragma unroll
    for (int r = 0; r < 8; ++r) {
      c0[r] = C[(size_t)(p0 + r + mo) * Npitch + (ntile << 4) + l15];
      c1[r] = C[(size_t)(p0 + 16 + r + mo) * Npitch + (ntile << 4) + l15];
    }
    // TDM prefetch of tap 0 weight fragments into this wave's LDS slot.
    tdm_load_w(W + (size_t)ntile * 512,
               (unsigned)(uintptr_t)&sW[(0 * 4 + wave) * (KCH * 512)],
               KCH, 128u, wStride);
  }

  for (int t = 0; t < ntaps; ++t) {
    const int dy = dyB + t / kw;
    const int dx = dxB + t % kw;
    const int hs = h + dy;
    const bool more = (t + 1) < ntaps;
    if (active && more) {
      tdm_load_w(W + ((size_t)(t + 1) * KCH * NT + ntile) * 512,
                 (unsigned)(uintptr_t)&sW[(((t + 1) & 1) * 4 + wave) * (KCH * 512)],
                 KCH, 128u, wStride);
    }
    const _Float16* wl = &sW[((t & 1) * 4 + wave) * (KCH * 512)];
#pragma unroll
    for (int kc = 0; kc < KCH; ++kc) {
      __syncthreads();
      // stage 32x32 f16 A tile, zero out-of-bounds (causal padding)
      for (int idx = tid << 2; idx < 1024; idx += (int)(blockDim.x << 2)) {
        int m  = idx >> 5;
        int ch = idx & 31;
        int wsm = w0 + m + dx;
        unsigned long long val = 0ull;
        if ((unsigned)hs < 64u && (unsigned)wsm < 64u) {
          const _Float16* src =
              A + (size_t)(nbase + (hs << 6) + wsm) * Apitch + kc * 32 + ch;
          val = *reinterpret_cast<const unsigned long long*>(src);
        }
        *reinterpret_cast<unsigned long long*>(&sA[idx]) = val;
      }
      __syncthreads();
      if (active) {
        if (kc == 0) {
          if (more) __builtin_amdgcn_s_wait_tensorcnt(1);
          else      __builtin_amdgcn_s_wait_tensorcnt(0);
        }
        const int kb = (lane >> 4) << 4;   // lanes16-31: K offset +16
        v8h a0l = *(const v8h*)&sA[(l15 << 5) + kb];
        v8h a0h = *(const v8h*)&sA[(l15 << 5) + kb + 8];
        v16h a0 = __builtin_shufflevector(a0l, a0h,
                    0,1,2,3,4,5,6,7,8,9,10,11,12,13,14,15);
        v8h a1l = *(const v8h*)&sA[((l15 + 16) << 5) + kb];
        v8h a1h = *(const v8h*)&sA[((l15 + 16) << 5) + kb + 8];
        v16h a1 = __builtin_shufflevector(a1l, a1h,
                    0,1,2,3,4,5,6,7,8,9,10,11,12,13,14,15);
        v16h b = *(const v16h*)&wl[kc * 512 + lane * 16];
        c0 = __builtin_amdgcn_wmma_f32_16x16x32_f16(
                 false, a0, false, b, (short)0, c0, false, false);
        c1 = __builtin_amdgcn_wmma_f32_16x16x32_f16(
                 false, a1, false, b, (short)0, c1, false, false);
      }
    }
  }

  if (active) {
#pragma unroll
    for (int r = 0; r < 8; ++r) {
      C[(size_t)(p0 + r + mo) * Npitch + (ntile << 4) + l15] = c0[r];
      C[(size_t)(p0 + 16 + r + mo) * Npitch + (ntile << 4) + l15] = c1[r];
    }
  }
}

// ---------------------------------------------------------------------------
// GatedActivation2d from the f32 accumulator: tanh(even) * sigmoid(odd)
// ---------------------------------------------------------------------------
__global__ void gate_f32(const float* __restrict__ C, _Float16* __restrict__ dst,
                         int halfN)
{
  size_t idx = (size_t)blockIdx.x * blockDim.x + threadIdx.x;
  size_t total = (size_t)NPIX * halfN;
  if (idx >= total) return;
  int j = (int)(idx % halfN);
  size_t p = idx / halfN;
  float a = C[p * (size_t)(halfN * 2) + 2 * j];
  float b = C[p * (size_t)(halfN * 2) + 2 * j + 1];
  float s = 1.f / (1.f + __expf(-b));
  dst[idx] = (_Float16)(tanhf(a) * s);
}

__global__ void cvt_store(const float* __restrict__ src, _Float16* __restrict__ dst,
                          size_t total, int relu)
{
  size_t i = (size_t)blockIdx.x * blockDim.x + threadIdx.x;
  if (i >= total) return;
  float v = src[i];
  if (relu) v = fmaxf(v, 0.f);
  dst[i] = (_Float16)v;
}

// ---------------------------------------------------------------------------
// Final 96 -> 768 masked GEMM, bias init in-register, scattered store into
// the (n, 256, C, h, w) output layout.
// ---------------------------------------------------------------------------
__global__ __launch_bounds__(128) void out2_gemm(
    const _Float16* __restrict__ A, const _Float16* __restrict__ W,
    const float* __restrict__ bias, float* __restrict__ out)
{
  __shared__ __align__(16) _Float16 sA[512];
  const int tid = threadIdx.x, lane = tid & 31, wave = tid >> 5;
  const int ntile = blockIdx.y * 4 + wave;     // NT = 48
  const int p0 = blockIdx.x << 4;
  const int l15 = lane & 15, mo = (lane >> 4) << 3;
  const int o = (ntile << 4) + l15;

  v8f c;
  float bv = bias[o];
#pragma unroll
  for (int r = 0; r < 8; ++r) c[r] = bv;

  for (int kc = 0; kc < 3; ++kc) {
    __syncthreads();
    {
      int idx = tid << 2;
      int m = idx >> 5, ch = idx & 31;
      *reinterpret_cast<unsigned long long*>(&sA[idx]) =
          *reinterpret_cast<const unsigned long long*>(
              A + (size_t)(p0 + m) * 96 + kc * 32 + ch);
    }
    __syncthreads();
    const int kb = (lane >> 4) << 4;
    v8h a0 = *(const v8h*)&sA[(l15 << 5) + kb];
    v8h a1 = *(const v8h*)&sA[(l15 << 5) + kb + 8];
    v16h a = __builtin_shufflevector(a0, a1,
               0,1,2,3,4,5,6,7,8,9,10,11,12,13,14,15);
    v16h b = *(const v16h*)(W + (((size_t)kc * 48 + ntile) << 9) + (lane << 4));
    c = __builtin_amdgcn_wmma_f32_16x16x32_f16(
            false, a, false, b, (short)0, c, false, false);
  }

  const int cc = o >> 8, q = o & 255;
#pragma unroll
  for (int r = 0; r < 8; ++r) {
    int p = p0 + r + mo;
    int nb = p >> 12, hw = p & 4095;
    out[(size_t)nb * 3145728 + (size_t)q * 12288 + (size_t)cc * 4096 + hw] = c[r];
  }
}

// ---------------------------------------------------------------------------
extern "C" void kernel_launch(void* const* d_in, const int* in_sizes, int n_in,
                              void* d_out, int out_size, void* d_ws, size_t ws_size,
                              hipStream_t stream)
{
  (void)in_sizes; (void)n_in; (void)out_size; (void)ws_size;
  const int NL = 16;
  const float* x = (const float*)d_in[0];
  const float *vconv_w[16], *vconv_b[16], *hconv_w[16], *hsub_w[16], *hconv_b[16],
              *v2h_w[16], *v2h_b[16], *hact_w[16], *hact_b[16];
  for (int i = 0; i < 16; ++i) {
    vconv_w[i] = (const float*)d_in[1 + i];
    vconv_b[i] = (const float*)d_in[17 + i];
    hconv_w[i] = (const float*)d_in[33 + i];
    hsub_w[i]  = (const float*)d_in[49 + i];
    hconv_b[i] = (const float*)d_in[65 + i];
    v2h_w[i]   = (const float*)d_in[81 + i];
    v2h_b[i]   = (const float*)d_in[97 + i];
    hact_w[i]  = (const float*)d_in[113 + i];
    hact_b[i]  = (const float*)d_in[129 + i];
  }
  const float* out1_w = (const float*)d_in[145];
  const float* out1_b = (const float*)d_in[146];
  const float* out2_w = (const float*)d_in[147];
  const float* out2_b = (const float*)d_in[148];

  // ---- workspace layout ----
  char* ws = (char*)d_ws;
  size_t off = 0;
  auto alloc = [&](size_t bytes) -> char* {
    off = (off + 255) & ~(size_t)255;
    char* p = ws + off;
    off += bytes;
    return p;
  };
  float*    Cbuf = (float*)   alloc((size_t)NPIX * 192 * sizeof(float));
  _Float16* vbuf = (_Float16*)alloc((size_t)NPIX * 192 * 2);
  _Float16* vxb  = (_Float16*)alloc((size_t)NPIX * 96 * 2);
  _Float16* hxA  = (_Float16*)alloc((size_t)NPIX * 96 * 2);
  _Float16* hxB  = (_Float16*)alloc((size_t)NPIX * 96 * 2);
  _Float16* gbuf = (_Float16*)alloc((size_t)NPIX * 96 * 2);

  _Float16 *Wv[16], *Wh[16], *Whs[16], *Wv2h[16], *Whact[16], *Wout1, *Wout2;
  for (int i = 0; i < NL; ++i) {
    int k = (i == 0) ? 7 : 3;
    int Kch = (i == 0) ? 1 : 3;
    Wv[i]    = (_Float16*)alloc((size_t)k * k * Kch * 12 * 512 * 2);
    Wh[i]    = (_Float16*)alloc((size_t)(k - 1) * Kch * 12 * 512 * 2);
    Whs[i]   = (_Float16*)alloc((size_t)Kch * 12 * 512 * 2);
    Wv2h[i]  = (_Float16*)alloc((size_t)6 * 12 * 512 * 2);
    Whact[i] = (_Float16*)alloc((size_t)3 * 6 * 512 * 2);
  }
  Wout1 = (_Float16*)alloc((size_t)3 * 6 * 512 * 2);
  Wout2 = (_Float16*)alloc((size_t)3 * 48 * 512 * 2);

  auto packLaunch = [&](const float* w, _Float16* dst, int O, int I, int taps,
                        int Kch, int NT, int mask) {
    int total = taps * Kch * NT * 32;
    pack_w<<<(total + 127) / 128, 128, 0, stream>>>(w, dst, O, I, taps, Kch, NT, mask);
  };
  for (int i = 0; i < NL; ++i) {
    int k = (i == 0) ? 7 : 3;
    int Iin = (i == 0) ? 4 : 96;
    int Kch = (i == 0) ? 1 : 3;
    packLaunch(vconv_w[i], Wv[i],   192, Iin, k * k, Kch, 12, 0);
    packLaunch(hconv_w[i], Wh[i],   192, Iin, k - 1, Kch, 12, 0);
    packLaunch(hsub_w[i],  Whs[i],  192, Iin, 1,     Kch, 12, (i == 0) ? 1 : 2);
    packLaunch(v2h_w[i],   Wv2h[i], 192, 192, 1,     6,   12, 0);
    packLaunch(hact_w[i],  Whact[i], 96,  96, 1,     3,    6, 2);
  }
  packLaunch(out1_w, Wout1,  96, 96, 1, 3,  6, 2);
  packLaunch(out2_w, Wout2, 768, 96, 1, 3, 48, 2);

  prep_input<<<NPIX / 256, 256, 0, stream>>>(x, hxA, vxb);

  const int g192 = (NPIX * 192 + 255) / 256;
  const int g96  = (NPIX * 96 + 255) / 256;

  auto launchGemm = [&](const _Float16* A, int Apitch, int Kch, const _Float16* W,
                        int NT, int ntaps, int kw, int dyB, int dxB,
                        float* C, int Npitch, int wpb) {
    dim3 grid(2048, NT / wpb);
    int thr = wpb * 32;
    switch (Kch) {
      case 1: gemm_tap<1><<<grid, thr, 0, stream>>>(A, Apitch, W, NT, ntaps, kw,
                                                    dyB, dxB, C, Npitch); break;
      case 3: gemm_tap<3><<<grid, thr, 0, stream>>>(A, Apitch, W, NT, ntaps, kw,
                                                    dyB, dxB, C, Npitch); break;
      default: gemm_tap<6><<<grid, thr, 0, stream>>>(A, Apitch, W, NT, ntaps, kw,
                                                     dyB, dxB, C, Npitch); break;
    }
  };

  _Float16* hx_in = hxA;
  _Float16* hx_out = hxB;
  int hxPitch = 32;
  for (int i = 0; i < NL; ++i) {
    int k = (i == 0) ? 7 : 3;
    int Kch = (i == 0) ? 1 : 3;
    int vxPitch = (i == 0) ? 32 : 96;
    // ---- vertical stack: v = conv(vx) + b ; vx' = gated(v) ----
    init_c<<<g192, 256, 0, stream>>>(Cbuf, vconv_b[i], nullptr, nullptr, 192, 192);
    launchGemm(vxb, vxPitch, Kch, Wv[i], 12, k * k, k, -(k - 1), -(k / 2), Cbuf, 192, 4);
    cvt_store<<<g192, 256, 0, stream>>>(Cbuf, vbuf, (size_t)NPIX * 192, 0);
    gate_f32<<<g96, 256, 0, stream>>>(Cbuf, vxb, 96);
    // ---- horizontal stack: hc = hconv(hx) + hsub@hx + v2h@v + biases ----
    init_c<<<g192, 256, 0, stream>>>(Cbuf, hconv_b[i], v2h_b[i], nullptr, 192, 192);
    launchGemm(hx_in, hxPitch, Kch, Wh[i], 12, k - 1, k - 1, 0, -(k - 1), Cbuf, 192, 4);
    launchGemm(hx_in, hxPitch, Kch, Whs[i], 12, 1, 1, 0, 0, Cbuf, 192, 4);
    launchGemm(vbuf, 192, 6, Wv2h[i], 12, 1, 1, 0, 0, Cbuf, 192, 4);
    gate_f32<<<g96, 256, 0, stream>>>(Cbuf, gbuf, 96);
    // ---- hx' = hact@g + b + residual ----
    init_c<<<g96, 256, 0, stream>>>(Cbuf, hact_b[i], nullptr,
                                    (i > 0) ? hx_in : nullptr, 96, 96);
    launchGemm(gbuf, 96, 3, Whact[i], 6, 1, 1, 0, 0, Cbuf, 96, 2);
    cvt_store<<<g96, 256, 0, stream>>>(Cbuf, hx_out, (size_t)NPIX * 96, 0);
    _Float16* tmp = hx_in; hx_in = hx_out; hx_out = tmp;
    hxPitch = 96;
  }

  // ---- head: o = relu(out1@hx + b) ; logits = out2@o + b ----
  init_c<<<g96, 256, 0, stream>>>(Cbuf, out1_b, nullptr, nullptr, 96, 96);
  launchGemm(hx_in, 96, 3, Wout1, 6, 1, 1, 0, 0, Cbuf, 96, 2);
  cvt_store<<<g96, 256, 0, stream>>>(Cbuf, gbuf, (size_t)NPIX * 96, 1);
  out2_gemm<<<dim3(4096, 12), 128, 0, stream>>>(gbuf, Wout2, out2_b, (float*)d_out);
}
